// ResidualAttentionBlock_ToME_86620900426086
// MI455X (gfx1250) — matrix operand order repaired and checked by reading, not compiled, not run on hardware
//
#include <hip/hip_runtime.h>

typedef __bf16 bf16;
typedef __attribute__((ext_vector_type(16))) __bf16 v16bf;
typedef __attribute__((ext_vector_type(8)))  float  v8f;
typedef __attribute__((ext_vector_type(4)))  unsigned int u32x4;
typedef __attribute__((ext_vector_type(8)))  int i32x8;
typedef __attribute__((ext_vector_type(4)))  int i32x4;
typedef int v4i32 __attribute__((vector_size(16)));   // matches async-LDS builtin pointee

#define AS1 __attribute__((address_space(1)))
#define AS3 __attribute__((address_space(3)))

#define SEQ    1025
#define BATCH  16
#define DMODEL 768
#define NHEAD  12
#define HDIM   64
#define DQKV   2304
#define DFF    3072
#define NTOK   (SEQ*BATCH)        // 16400
#define RMERGE 128
#define NSRC   513
#define NDST   512
#define NUNM   (NSRC-RMERGE)      // 385
#define NEWN   (NUNM+NDST)        // 897
#define NTOK2  (NEWN*BATCH)       // 14352
#define NKT    65                 // ceil(1025/16)

#if __has_builtin(__builtin_amdgcn_global_load_async_to_lds_b128) && \
    __has_builtin(__builtin_amdgcn_s_wait_asynccnt)
#define HAVE_ASYNC_LDS 1
#else
#define HAVE_ASYNC_LDS 0
#endif

#if __has_builtin(__builtin_amdgcn_tensor_load_to_lds) && \
    __has_builtin(__builtin_amdgcn_s_wait_tensorcnt)
#define HAVE_TDM 1
#else
#define HAVE_TDM 0
#endif

__device__ __forceinline__ v8f wmma_bf16(v16bf a, v16bf b, v8f c) {
  return __builtin_amdgcn_wmma_f32_16x16x32_bf16(false, a, false, b, (short)0, c, false, false);
}

__device__ __forceinline__ unsigned lds_addr_of(void* p) {
  return (unsigned)(unsigned long long)(AS3 char*)p;
}

#if HAVE_ASYNC_LDS
__device__ __forceinline__ void async_copy_b128(const void* g, void* l) {
  __builtin_amdgcn_global_load_async_to_lds_b128(
      (AS1 v4i32*)(void*)g, (AS3 v4i32*)l, 0, 0);
}
#endif

#if HAVE_TDM
// Build a D# for a 2D tile (elems_x 2-byte elements per row, rows_tile rows,
// row stride stride_elems). Rows >= rows_valid are zero-filled by TDM OOB.
__device__ __forceinline__ void tdm_load_2d_to_lds(
    const void* gptr, unsigned lds_off, unsigned elems_x, unsigned rows_tile,
    unsigned rows_valid, unsigned long long stride_elems) {
  unsigned long long ga = (unsigned long long)gptr;
  u32x4 g0;
  g0[0] = 1u;                                                   // count=1
  g0[1] = lds_off;                                              // lds_addr
  g0[2] = (unsigned)(ga & 0xFFFFFFFFu);                         // global_addr lo
  g0[3] = (unsigned)((ga >> 32) & 0x01FFFFFFu) | 0x80000000u;   // hi | type=2
  i32x8 g1;
  g1[0] = (int)(1u << 16);                                      // data_size=2B
  g1[1] = (int)((elems_x & 0xFFFFu) << 16);                     // tensor_dim0 lo16
  g1[2] = (int)((elems_x >> 16) | ((rows_valid & 0xFFFFu) << 16)); // dim0 hi | dim1 lo
  g1[3] = (int)((rows_valid >> 16) | ((elems_x & 0xFFFFu) << 16)); // dim1 hi | tile_dim0
  g1[4] = (int)(rows_tile & 0xFFFFu);                           // tile_dim1, tile_dim2=0
  g1[5] = (int)(stride_elems & 0xFFFFFFFFu);                    // dim0_stride lo32
  g1[6] = (int)((stride_elems >> 32) & 0xFFFFu);                // dim0_stride hi16
  g1[7] = 0;
  i32x4 z4; z4[0] = z4[1] = z4[2] = z4[3] = 0;
#if __clang_major__ >= 23
  i32x8 z8;
#pragma unroll
  for (int i = 0; i < 8; i++) z8[i] = 0;
  __builtin_amdgcn_tensor_load_to_lds(g0, g1, z4, z4, z8, 0);
#else
  __builtin_amdgcn_tensor_load_to_lds(g0, g1, z4, z4, 0);
#endif
}
#endif

// ---------------------------------------------------------------- utilities
__global__ __launch_bounds__(256) void zero_f32(float* __restrict__ p, int n) {
  int i = blockIdx.x * 256 + threadIdx.x;
  if (i < n) p[i] = 0.f;
}

__global__ __launch_bounds__(256) void f32_to_bf16(const float* __restrict__ in,
                                                   bf16* __restrict__ out, int n) {
  int i = blockIdx.x * 256 + threadIdx.x;
  if (i < n) out[i] = (bf16)in[i];
}

// ------------------------------------------------------- layernorm -> bf16
__global__ __launch_bounds__(256) void layernorm_to_bf16(
    const float* __restrict__ x, const float* __restrict__ g,
    const float* __restrict__ bt, bf16* __restrict__ out, int D) {
  int row = blockIdx.x;
  const float* xr = x + (size_t)row * D;
  __shared__ float red[256];
  int tid = threadIdx.x;

  float s = 0.f;
  for (int d = tid; d < D; d += 256) s += xr[d];
  red[tid] = s; __syncthreads();
  for (int o = 128; o > 0; o >>= 1) { if (tid < o) red[tid] += red[tid + o]; __syncthreads(); }
  float mu = red[0] / D; __syncthreads();

  float v = 0.f;
  for (int d = tid; d < D; d += 256) { float t = xr[d] - mu; v += t * t; }
  red[tid] = v; __syncthreads();
  for (int o = 128; o > 0; o >>= 1) { if (tid < o) red[tid] += red[tid + o]; __syncthreads(); }
  float rs = rsqrtf(red[0] / D + 1e-5f);

  for (int d = tid; d < D; d += 256)
    out[(size_t)row * D + d] = (bf16)((xr[d] - mu) * rs * g[d] + bt[d]);
}

// --------------------------------------------------------- WMMA bf16 GEMM
// C[M,N] = A[M,K] @ W[N,K]^T + bias. 8 waves/block share one 16-row M tile;
// the A tile is staged through LDS (async load-to-LDS when available) so each
// block fetches A once instead of 8x. Per-wave 16x16 output tile.
// EP: 0 = bf16 out, 1 = f32 out + residual, 2 = swish(1.702) -> bf16 out
#define KCH 256
template<int EP>
__global__ __launch_bounds__(256) void gemm_bf16_wmma(
    const bf16* __restrict__ A, const bf16* __restrict__ W,
    const float* __restrict__ bias, const float* __restrict__ res,
    void* __restrict__ out, int N, int K) {
  int lane = threadIdx.x & 31;
  int wave = threadIdx.x >> 5;
  int lm = lane & 15, lh = lane >> 4;
  int mt = blockIdx.y;
  int nt = blockIdx.x * 8 + wave;

  __shared__ __align__(16) bf16 lds_a[16][KCH];

  const bf16* bp = W + (size_t)(nt * 16 + lm) * K;

  v8f acc;
#pragma unroll
  for (int r = 0; r < 8; r++) acc[r] = 0.f;

  for (int k0 = 0; k0 < K; k0 += KCH) {
    // stage 16 x KCH bf16 A tile (8KB) cooperatively: 512 x 16B transfers
#pragma unroll
    for (int it = 0; it < 2; it++) {
      int id = it * 256 + threadIdx.x;
      int row = id >> 5;
      int col = (id & 31) << 3;
      const bf16* gp = A + (size_t)(mt * 16 + row) * K + k0 + col;
#if HAVE_ASYNC_LDS
      async_copy_b128(gp, &lds_a[row][col]);
#else
      *(float4*)&lds_a[row][col] = *(const float4*)gp;
#endif
    }
#if HAVE_ASYNC_LDS
    __builtin_amdgcn_s_wait_asynccnt(0);
#endif
    __syncthreads();

    for (int kk = 0; kk < KCH; kk += 32) {
      v16bf a, b;
      int ka = kk + lh * 16;
#pragma unroll
      for (int i = 0; i < 16; i++) a[i] = lds_a[lm][ka + i];
      int kb = k0 + kk + lh * 16;
#pragma unroll
      for (int i = 0; i < 16; i++) b[i] = bp[kb + i];
      acc = wmma_bf16(a, b, acc);
    }
    __syncthreads();
  }

  int col = nt * 16 + lm;
  float bv = bias[col];
#pragma unroll
  for (int r = 0; r < 8; r++) {
    int row = mt * 16 + lh * 8 + r;
    size_t o = (size_t)row * N + col;
    float v = acc[r] + bv;
    if (EP == 1) {
      ((float*)out)[o] = v + res[o];
    } else if (EP == 2) {
      float sg = 1.f / (1.f + __expf(-1.702f * v));
      ((bf16*)out)[o] = (bf16)(v * sg);
    } else {
      ((bf16*)out)[o] = (bf16)v;
    }
  }
}

// --------------------------------------------------- attention (flash 2-pass)
// One wave per (qtile, head, batch). Accumulates mean attention via atomics.
// V tiles are staged into LDS by the Tensor Data Mover when available.
__global__ __launch_bounds__(32) void attention_kernel(
    const bf16* __restrict__ qkv, float* __restrict__ meanw, bf16* __restrict__ ctx) {
  const int b = blockIdx.z, h = blockIdx.y, mt = blockIdx.x;
  const int lane = threadIdx.x;
  const int lm = lane & 15, lh = lane >> 4;
  const float scale = 0.125f;       // 1/sqrt(64)
  const float hinv  = 1.f / (float)NHEAD;

  // Q fragments (16 rows x 64 K as two 16x32 A-fragments)
  int qr = mt * 16 + lm; if (qr > SEQ - 1) qr = SEQ - 1;
  const bf16* qp = qkv + (size_t)(qr * BATCH + b) * DQKV + h * HDIM;
  v16bf qa[2];
#pragma unroll
  for (int kc = 0; kc < 2; kc++) {
    int ka = kc * 32 + lh * 16;
#pragma unroll
    for (int i = 0; i < 16; i++) qa[kc][i] = qp[ka + i];
  }

  float mrun[8], lrun[8];
#pragma unroll
  for (int r = 0; r < 8; r++) { mrun[r] = -__builtin_inff(); lrun[r] = 0.f; }

  // ---- pass A: row max + sum(exp)
  for (int jt = 0; jt < NKT; jt++) {
    int key = jt * 16 + lm;
    int keyc = key > SEQ - 1 ? SEQ - 1 : key;
    const bf16* kp = qkv + (size_t)(keyc * BATCH + b) * DQKV + DMODEL + h * HDIM;
    v8f s;
#pragma unroll
    for (int r = 0; r < 8; r++) s[r] = 0.f;
#pragma unroll
    for (int kc = 0; kc < 2; kc++) {
      v16bf kb;
#pragma unroll
      for (int i = 0; i < 16; i++) kb[i] = kp[kc * 32 + lh * 16 + i];
      s = wmma_bf16(qa[kc], kb, s);
    }
#pragma unroll
    for (int r = 0; r < 8; r++) {
      float v = (key < SEQ) ? s[r] * scale : -__builtin_inff();
      float tm = v;
      tm = fmaxf(tm, __shfl_xor(tm, 1, 32));
      tm = fmaxf(tm, __shfl_xor(tm, 2, 32));
      tm = fmaxf(tm, __shfl_xor(tm, 4, 32));
      tm = fmaxf(tm, __shfl_xor(tm, 8, 32));
      float mnew = fmaxf(mrun[r], tm);
      float p = __expf(v - mnew);
      p += __shfl_xor(p, 1, 32);
      p += __shfl_xor(p, 2, 32);
      p += __shfl_xor(p, 4, 32);
      p += __shfl_xor(p, 8, 32);
      lrun[r] = lrun[r] * __expf(mrun[r] - mnew) + p;
      mrun[r] = mnew;
    }
  }

  float linv[8];
#pragma unroll
  for (int r = 0; r < 8; r++) linv[r] = (lrun[r] > 0.f) ? 1.f / lrun[r] : 0.f;

  // ---- pass B: P = softmax, accumulate P@V with WMMA, P/H into mean buffer
  __shared__ __align__(16) bf16 lds_p[16][32];
  __shared__ __align__(16) bf16 lds_v[32][HDIM];
  v8f acc[4];
#pragma unroll
  for (int c = 0; c < 4; c++)
#pragma unroll
    for (int r = 0; r < 8; r++) acc[c][r] = 0.f;

  for (int jp = 0; jp < (NKT + 1) / 2; jp++) {
    // stage 32 V rows (this head) into LDS
    int kv0 = jp * 32;
#if HAVE_TDM
    tdm_load_2d_to_lds(qkv + (size_t)(kv0 * BATCH + b) * DQKV + 2 * DMODEL + h * HDIM,
                       lds_addr_of(&lds_v[0][0]), HDIM, 32,
                       (unsigned)(SEQ - kv0), (unsigned long long)BATCH * DQKV);
#else
    {
      int kv = kv0 + lane;
      int kvc = kv > SEQ - 1 ? SEQ - 1 : kv;
      bool vok = kv < SEQ;
      const bf16* vp = qkv + (size_t)(kvc * BATCH + b) * DQKV + 2 * DMODEL + h * HDIM;
#pragma unroll
      for (int i = 0; i < HDIM; i++) lds_v[lane][i] = vok ? vp[i] : (bf16)0.f;
    }
#endif
#pragma unroll
    for (int sub = 0; sub < 2; sub++) {
      int jt = jp * 2 + sub;
      float pv[8];
      if (jt < NKT) {
        int key = jt * 16 + lm;
        int keyc = key > SEQ - 1 ? SEQ - 1 : key;
        const bf16* kp = qkv + (size_t)(keyc * BATCH + b) * DQKV + DMODEL + h * HDIM;
        v8f s;
#pragma unroll
        for (int r = 0; r < 8; r++) s[r] = 0.f;
#pragma unroll
        for (int kc = 0; kc < 2; kc++) {
          v16bf kb;
#pragma unroll
          for (int i = 0; i < 16; i++) kb[i] = kp[kc * 32 + lh * 16 + i];
          s = wmma_bf16(qa[kc], kb, s);
        }
#pragma unroll
        for (int r = 0; r < 8; r++) {
          float v = (key < SEQ) ? s[r] * scale : -__builtin_inff();
          pv[r] = __expf(v - mrun[r]) * linv[r];
          int qrow = mt * 16 + lh * 8 + r;
          if (key < SEQ && qrow < SEQ)
            atomicAdd(&meanw[((size_t)b * SEQ + qrow) * SEQ + key], pv[r] * hinv);
        }
      } else {
#pragma unroll
        for (int r = 0; r < 8; r++) pv[r] = 0.f;
      }
#pragma unroll
      for (int r = 0; r < 8; r++) lds_p[lh * 8 + r][sub * 16 + lm] = (bf16)pv[r];
    }
#if HAVE_TDM
    __builtin_amdgcn_s_wait_tensorcnt(0);
#endif
    __syncthreads();
    // P as A-fragment (16x32), V as B-fragments (32x16 per hd chunk)
    v16bf pa;
#pragma unroll
    for (int i = 0; i < 16; i++) pa[i] = lds_p[lm][lh * 16 + i];
#pragma unroll
    for (int c = 0; c < 4; c++) {
      v16bf vb;
#pragma unroll
      for (int i = 0; i < 16; i++) vb[i] = lds_v[lh * 16 + i][c * 16 + lm];
      acc[c] = wmma_bf16(pa, vb, acc[c]);
    }
    __syncthreads();
  }

#pragma unroll
  for (int c = 0; c < 4; c++)
#pragma unroll
    for (int r = 0; r < 8; r++) {
      int qrow = mt * 16 + lh * 8 + r;
      if (qrow < SEQ)
        ctx[(size_t)(qrow * BATCH + b) * DMODEL + h * HDIM + c * 16 + lm] = (bf16)acc[c][r];
    }
}

// ------------------------------------------------- ToMe: metric row norms
__global__ __launch_bounds__(256) void row_norm(const float* __restrict__ mean,
                                                float* __restrict__ norms) {
  int row = blockIdx.x;                       // b*SEQ + n
  const float* rp = mean + (size_t)row * SEQ;
  __shared__ float red[256];
  int tid = threadIdx.x;
  float s = 0.f;
  for (int k = tid; k < SEQ; k += 256) { float v = rp[k]; s += v * v; }
  red[tid] = s; __syncthreads();
  for (int o = 128; o > 0; o >>= 1) { if (tid < o) red[tid] += red[tid + o]; __syncthreads(); }
  if (tid == 0) norms[row] = sqrtf(red[0]);
}

// ---------------------------------- ToMe: cosine scores + per-row argmax
__global__ __launch_bounds__(256) void bip_scores_argmax(
    const float* __restrict__ mean, const float* __restrict__ norms,
    float* __restrict__ node_max, int* __restrict__ node_idx) {
  int b = blockIdx.y;
  int i = blockIdx.x;                        // src row index 0..512
  int arow = 2 * i;
  __shared__ float sa[SEQ];
  __shared__ float rv[256];
  __shared__ int   ri[256];
  int tid = threadIdx.x;

  const float* ap = mean + ((size_t)b * SEQ + arow) * SEQ;
  for (int k = tid; k < SEQ; k += 256) sa[k] = ap[k];
  __syncthreads();
  float na = norms[b * SEQ + arow];

  float best = -__builtin_inff(); int bestj = 0;
  for (int j = tid; j < NDST; j += 256) {
    int brow = 2 * j + 1;
    const float* bp = mean + ((size_t)b * SEQ + brow) * SEQ;
    float dot = 0.f;
    for (int k = 0; k < SEQ; k++) dot += sa[k] * bp[k];
    float sc = dot / (na * norms[b * SEQ + brow]);
    if (sc > best) { best = sc; bestj = j; }
  }
  rv[tid] = best; ri[tid] = bestj; __syncthreads();
  for (int o = 128; o > 0; o >>= 1) {
    if (tid < o) {
      float ov = rv[tid + o]; int oj = ri[tid + o];
      if (ov > rv[tid] || (ov == rv[tid] && oj < ri[tid])) { rv[tid] = ov; ri[tid] = oj; }
    }
    __syncthreads();
  }
  if (tid == 0) {
    node_max[b * NSRC + i] = (i == 0) ? -__builtin_inff() : rv[0];
    node_idx[b * NSRC + i] = (i == 0) ? 0 : ri[0];
  }
}

// -------------------------------------- ToMe: sort + select merge indices
__global__ __launch_bounds__(512) void tome_sort(
    const float* __restrict__ node_max, const int* __restrict__ node_idx,
    int* __restrict__ unm_idx, int* __restrict__ src_idx, int* __restrict__ dst_idx) {
  int b = blockIdx.x;
  int tid = threadIdx.x;
  __shared__ float sv[1024];
  __shared__ int   si[1024];
  __shared__ int   s2[512];

  for (int t = tid; t < 1024; t += 512) {
    if (t < NSRC) { sv[t] = node_max[b * NSRC + t]; si[t] = t; }
    else          { sv[t] = -__builtin_inff();      si[t] = 100000 + t; }
  }
  __syncthreads();

  // bitonic sort: descending by value, ties ascending by index (stable argsort)
  for (int k = 2; k <= 1024; k <<= 1)
    for (int j = k >> 1; j > 0; j >>= 1) {
      for (int t = tid; t < 1024; t += 512) {
        int ixj = t ^ j;
        if (ixj > t) {
          bool up = ((t & k) == 0);
          float v1 = sv[t], v2 = sv[ixj];
          int i1 = si[t], i2 = si[ixj];
          bool first = (v1 > v2) || (v1 == v2 && i1 < i2);
          bool ok = up ? first : !first;
          if (!ok) { sv[t] = v2; sv[ixj] = v1; si[t] = i2; si[ixj] = i1; }
        }
      }
      __syncthreads();
    }

  if (tid < RMERGE) {
    int s = si[tid];
    src_idx[b * RMERGE + tid] = s;
    dst_idx[b * RMERGE + tid] = node_idx[b * NSRC + s];
  }
  s2[tid] = (tid < NUNM) ? si[RMERGE + tid] : 0x7FFFFFFF;
  __syncthreads();

  // ascending int sort of the unmerged indices
  for (int k = 2; k <= 512; k <<= 1)
    for (int j = k >> 1; j > 0; j >>= 1) {
      int t = tid, ixj = t ^ j;
      if (ixj > t) {
        int a = s2[t], c = s2[ixj];
        bool up = ((t & k) == 0);
        bool swp = up ? (a > c) : (a < c);
        if (swp) { s2[t] = c; s2[ixj] = a; }
      }
      __syncthreads();
    }
  if (tid < NUNM) unm_idx[b * NUNM + tid] = s2[tid];
}

// ------------------------------------------------------------ token merge
__global__ __launch_bounds__(256) void merge_copy(
    const float* __restrict__ x1, const int* __restrict__ unm_idx,
    float* __restrict__ newx, float* __restrict__ sz) {
  int row = blockIdx.x;              // t*BATCH + b
  int t = row >> 4, b = row & 15;
  int n = (t < NUNM) ? 2 * unm_idx[b * NUNM + t] : 2 * (t - NUNM) + 1;
  const float* src = x1 + (size_t)(n * BATCH + b) * DMODEL;
  float* dst = newx + (size_t)row * DMODEL;
  for (int d = threadIdx.x; d < DMODEL; d += 256) dst[d] = src[d];
  if (threadIdx.x == 0) sz[row] = 1.f;
}

__global__ __launch_bounds__(256) void merge_scatter(
    const float* __restrict__ x1, const int* __restrict__ src_idx,
    const int* __restrict__ dst_idx, float* __restrict__ newx, float* __restrict__ sz) {
  int s = blockIdx.x, b = blockIdx.y;
  int n = 2 * src_idx[b * RMERGE + s];
  int t = NUNM + dst_idx[b * RMERGE + s];
  const float* sp = x1 + (size_t)(n * BATCH + b) * DMODEL;
  float* dp = newx + (size_t)(t * BATCH + b) * DMODEL;
  for (int d = threadIdx.x; d < DMODEL; d += 256) atomicAdd(&dp[d], sp[d]);
  if (threadIdx.x == 0) atomicAdd(&sz[t * BATCH + b], 1.f);
}

__global__ __launch_bounds__(256) void divide_and_size(
    float* __restrict__ newx, const float* __restrict__ sz, float* __restrict__ out_size) {
  int row = blockIdx.x;
  float s = sz[row];
  float inv = 1.f / s;
  for (int d = threadIdx.x; d < DMODEL; d += 256) newx[(size_t)row * DMODEL + d] *= inv;
  if (threadIdx.x == 0) out_size[row] = s;
}

// ---------------------------------------------------------------- launcher
extern "C" void kernel_launch(void* const* d_in, const int* in_sizes, int n_in,
                              void* d_out, int out_size, void* d_ws, size_t ws_size,
                              hipStream_t stream) {
  (void)in_sizes; (void)n_in; (void)out_size; (void)ws_size;

  const float* x      = (const float*)d_in[0];
  const float* in_w   = (const float*)d_in[1];
  const float* in_b   = (const float*)d_in[2];
  const float* outp_w = (const float*)d_in[3];
  const float* outp_b = (const float*)d_in[4];
  const float* ln1_g  = (const float*)d_in[5];
  const float* ln1_b  = (const float*)d_in[6];
  const float* ln2_g  = (const float*)d_in[7];
  const float* ln2_b  = (const float*)d_in[8];
  const float* fc_w   = (const float*)d_in[9];
  const float* fc_b   = (const float*)d_in[10];
  const float* proj_w = (const float*)d_in[11];
  const float* proj_b = (const float*)d_in[12];
  float* out = (float*)d_out;
  float* out_sz = out + (size_t)NTOK2 * DMODEL;

  // workspace bump allocator (256B aligned)
  char* p = (char*)d_ws;
  auto bump = [&](size_t bytes) -> char* {
    char* r = p; p += (bytes + 255) & ~(size_t)255; return r;
  };
  bf16* wqkv_bf = (bf16*)bump((size_t)DQKV * DMODEL * 2);
  bf16* wout_bf = (bf16*)bump((size_t)DMODEL * DMODEL * 2);
  bf16* wfc_bf  = (bf16*)bump((size_t)DFF * DMODEL * 2);
  bf16* wprj_bf = (bf16*)bump((size_t)DMODEL * DFF * 2);
  bf16* xn_bf   = (bf16*)bump((size_t)NTOK * DMODEL * 2);
  bf16* qkv_bf  = (bf16*)bump((size_t)NTOK * DQKV * 2);
  float* meanw  = (float*)bump((size_t)BATCH * SEQ * SEQ * 4);
  bf16* ctx_bf  = (bf16*)bump((size_t)NTOK * DMODEL * 2);
  float* x1     = (float*)bump((size_t)NTOK * DMODEL * 4);
  float* norms  = (float*)bump((size_t)NTOK * 4);
  float* nmax   = (float*)bump((size_t)BATCH * NSRC * 4);
  int*   nidx   = (int*)bump((size_t)BATCH * NSRC * 4);
  int*   unm    = (int*)bump((size_t)BATCH * NUNM * 4);
  int*   srci   = (int*)bump((size_t)BATCH * RMERGE * 4);
  int*   dsti   = (int*)bump((size_t)BATCH * RMERGE * 4);
  float* newx   = (float*)bump((size_t)NTOK2 * DMODEL * 4);
  float* sz     = (float*)bump((size_t)NTOK2 * 4);
  bf16* hin_bf  = (bf16*)bump((size_t)NTOK2 * DMODEL * 2);
  bf16* h1_bf   = (bf16*)bump((size_t)NTOK2 * DFF * 2);

  auto cv = [&](const float* src, bf16* dst, int n) {
    f32_to_bf16<<<(n + 255) / 256, 256, 0, stream>>>(src, dst, n);
  };
  cv(in_w,   wqkv_bf, DQKV * DMODEL);
  cv(outp_w, wout_bf, DMODEL * DMODEL);
  cv(fc_w,   wfc_bf,  DFF * DMODEL);
  cv(proj_w, wprj_bf, DMODEL * DFF);

  {
    int n = BATCH * SEQ * SEQ;
    zero_f32<<<(n + 255) / 256, 256, 0, stream>>>(meanw, n);
  }

  // LN1 -> bf16
  layernorm_to_bf16<<<NTOK, 256, 0, stream>>>(x, ln1_g, ln1_b, xn_bf, DMODEL);

  // QKV GEMM: [16400,768] x [2304,768]^T -> bf16
  gemm_bf16_wmma<0><<<dim3(DQKV / 128, NTOK / 16), 256, 0, stream>>>(
      xn_bf, wqkv_bf, in_b, nullptr, qkv_bf, DQKV, DMODEL);

  // attention (+ head-mean accumulation)
  attention_kernel<<<dim3(NKT, NHEAD, BATCH), 32, 0, stream>>>(qkv_bf, meanw, ctx_bf);

  // out-proj GEMM + residual x -> x1
  gemm_bf16_wmma<1><<<dim3(DMODEL / 128, NTOK / 16), 256, 0, stream>>>(
      ctx_bf, wout_bf, outp_b, x, x1, DMODEL, DMODEL);

  // ToMe bipartite matching on mean attention
  row_norm<<<NTOK, 256, 0, stream>>>(meanw, norms);
  bip_scores_argmax<<<dim3(NSRC, BATCH), 256, 0, stream>>>(meanw, norms, nmax, nidx);
  tome_sort<<<BATCH, 512, 0, stream>>>(nmax, nidx, unm, srci, dsti);

  // merge
  merge_copy<<<NTOK2, 256, 0, stream>>>(x1, unm, newx, sz);
  merge_scatter<<<dim3(RMERGE, BATCH), 256, 0, stream>>>(x1, srci, dsti, newx, sz);
  divide_and_size<<<NTOK2, 256, 0, stream>>>(newx, sz, out_sz);

  // LN2 -> bf16
  layernorm_to_bf16<<<NTOK2, 256, 0, stream>>>(newx, ln2_g, ln2_b, hin_bf, DMODEL);

  // FC GEMM + swish -> bf16
  gemm_bf16_wmma<2><<<dim3(DFF / 128, NTOK2 / 16), 256, 0, stream>>>(
      hin_bf, wfc_bf, fc_b, nullptr, h1_bf, DFF, DMODEL);

  // proj GEMM + residual newx -> d_out
  gemm_bf16_wmma<1><<<dim3(DMODEL / 128, NTOK2 / 16), 256, 0, stream>>>(
      h1_bf, wprj_bf, proj_b, newx, out, DMODEL, DFF);
}
